// GraphRNNDecoder_12275016532224
// MI455X (gfx1250) — compile-verified, matrix operand order checked by reading, not run on hardware
//
#include <hip/hip_runtime.h>
#include <hip/hip_bf16.h>

typedef __attribute__((ext_vector_type(16))) _Float16 v16h;
typedef __attribute__((ext_vector_type(8)))  float    v8f;

#define Bb   8
#define Tt   10
#define Nn   50
#define INN  6
#define Hh   256
#define Kk   4
#define Ee   (Nn * (Nn - 1))   /* 2450 */
#define BE   (Bb * Ee)         /* 19600 */

// Branch-free activations built on v_exp_f32 / v_rcp_f32 (TRANS ops that can
// co-execute with WMMA). Inputs here are small (weights ~N(0,0.05)), clamp
// keeps exp finite.
__device__ __forceinline__ float fast_tanh(float x) {
    x = fminf(15.0f, fmaxf(-15.0f, x));
    float e = __expf(2.0f * x);
    return (e - 1.0f) * __builtin_amdgcn_rcpf(e + 1.0f);
}
__device__ __forceinline__ float fast_sigmoid(float x) {
    float e = __expf(-x);               // overflow -> inf -> rcp -> 0 (correct)
    return __builtin_amdgcn_rcpf(1.0f + e);
}

// ---------------------------------------------------------------------------
// Init: hidden = 0 (f32 + f16), ins = inputs[:, 0]
// ---------------------------------------------------------------------------
__global__ void k_init(const float* __restrict__ inputs,
                       float* __restrict__ hidden,
                       _Float16* __restrict__ h16,
                       float* __restrict__ insbuf) {
    int i = blockIdx.x * blockDim.x + threadIdx.x;
    if (i < Bb * Nn * Hh) {
        hidden[i] = 0.0f;
        h16[i] = (_Float16)0.0f;
    }
    if (i < Bb * Nn * INN) {
        int b = i / (Nn * INN);
        int rem = i - b * (Nn * INN);
        insbuf[i] = inputs[(size_t)b * Tt * Nn * INN + rem];   // inputs[b][0][n][in]
    }
}

// ---------------------------------------------------------------------------
// Convert message weights to f16 (layout unchanged: B[k][n] = W[n][k] rows)
// ---------------------------------------------------------------------------
__global__ void k_cvtw(const float* __restrict__ w1, const float* __restrict__ w2,
                       _Float16* __restrict__ w1h, _Float16* __restrict__ w2h) {
    int i = blockIdx.x * blockDim.x + threadIdx.x;
    if (i < Kk * Hh * 2 * Hh) w1h[i] = (_Float16)w1[i];
    if (i < Kk * Hh * Hh)     w2h[i] = (_Float16)w2[i];
}

// ---------------------------------------------------------------------------
// Fused edge-message kernel:
//   all_msgs[r] = sum_{et=1..3} tanh(tanh(pre @ w1[et]^T + b1) @ w2[et]^T + b2)
//                 * rel[r][et] / 3
// pre tile (16 rows x 512) staged once in LDS; m1 tile staged in LDS between
// the two WMMA layers. Block: 256 threads = 8 waves; wave w owns col tiles
// {2w, 2w+1} (16 cols each) of the 256-wide output.
// ---------------------------------------------------------------------------
__global__ __launch_bounds__(256)
void k_msg(const _Float16* __restrict__ h16,
           const _Float16* __restrict__ w1h,
           const float* __restrict__ b1,
           const _Float16* __restrict__ w2h,
           const float* __restrict__ b2,
           const float* __restrict__ rel,   // sampled_edges [B,T,E,K]
           int t,
           float* __restrict__ all_msgs) {  // [BE, H]
    __shared__ _Float16 preS[16][520];      // 16 rows x 512 (+8 pad)
    __shared__ _Float16 m1s[16][264];       // 16 rows x 256 (+8 pad)

    const int tid   = threadIdx.x;
    const int lane  = tid & 31;
    const int wv    = tid >> 5;
    const int lhalf = lane >> 4;            // 0/1
    const int lmod  = lane & 15;
    const int r0    = blockIdx.x * 16;
    const int ct0   = wv * 2;

    // ---- stage pre tile: 16 threads per row, 32 contiguous halves each ----
    {
        int row  = tid >> 4;                // 0..15
        int seg  = tid & 15;                // 0..15
        int col0 = seg * 32;                // 0..480
        int r    = r0 + row;
        int b_   = r / Ee;
        int e_   = r - b_ * Ee;
        int snd_ = e_ / (Nn - 1);
        int c_   = e_ - snd_ * (Nn - 1);
        int rcv_ = c_ + ((c_ >= snd_) ? 1 : 0);
        const _Float16* src = (col0 < Hh)
            ? (h16 + (size_t)(b_ * Nn + rcv_) * Hh + col0)
            : (h16 + (size_t)(b_ * Nn + snd_) * Hh + (col0 - Hh));
#pragma unroll
        for (int j = 0; j < 32; ++j) preS[row][col0 + j] = src[j];
    }

    v8f accAll0 = {};
    v8f accAll1 = {};

    for (int et = 1; et < Kk; ++et) {
        __syncthreads();    // preS ready (1st iter) / m1s reads done (later)
        // ---------------- layer 1: K = 512, A from LDS ----------------
        v8f acc0 = {};
        v8f acc1 = {};
        const _Float16* w1base = w1h + (size_t)et * Hh * (2 * Hh);
        for (int k0 = 0; k0 < 2 * Hh; k0 += 32) {
            int ka = k0 + 8 * lhalf;
            const _Float16* pa = &preS[lmod][ka];
            v16h a;
#pragma unroll
            for (int j = 0; j < 8; ++j) { a[j] = pa[j]; a[8 + j] = pa[16 + j]; }
            {
                const _Float16* q = w1base + (size_t)(ct0 * 16 + lmod) * (2 * Hh) + ka;
                v16h bf;
#pragma unroll
                for (int j = 0; j < 8; ++j) { bf[j] = q[j]; bf[8 + j] = q[16 + j]; }
                acc0 = __builtin_amdgcn_wmma_f32_16x16x32_f16(false, a, false, bf,
                                                              (short)0, acc0, false, false);
            }
            {
                const _Float16* q = w1base + (size_t)((ct0 + 1) * 16 + lmod) * (2 * Hh) + ka;
                v16h bf;
#pragma unroll
                for (int j = 0; j < 8; ++j) { bf[j] = q[j]; bf[8 + j] = q[16 + j]; }
                acc1 = __builtin_amdgcn_wmma_f32_16x16x32_f16(false, a, false, bf,
                                                              (short)0, acc1, false, false);
            }
        }
        {
            float bia0 = b1[et * Hh + ct0 * 16 + lmod];
            float bia1 = b1[et * Hh + (ct0 + 1) * 16 + lmod];
#pragma unroll
            for (int j = 0; j < 8; ++j) {
                int m = j + 8 * lhalf;
                m1s[m][ct0 * 16 + lmod]       = (_Float16)fast_tanh(acc0[j] + bia0);
                m1s[m][(ct0 + 1) * 16 + lmod] = (_Float16)fast_tanh(acc1[j] + bia1);
            }
        }
        __syncthreads();
        // ---------------- layer 2: K = 256, A from LDS ----------------
        v8f acc2a = {};
        v8f acc2b = {};
        const _Float16* w2base = w2h + (size_t)et * Hh * Hh;
        for (int k0 = 0; k0 < Hh; k0 += 32) {
            int ka = k0 + 8 * lhalf;
            const _Float16* pa = &m1s[lmod][ka];
            v16h a2;
#pragma unroll
            for (int j = 0; j < 8; ++j) { a2[j] = pa[j]; a2[8 + j] = pa[16 + j]; }
            {
                const _Float16* q = w2base + (size_t)(ct0 * 16 + lmod) * Hh + ka;
                v16h bf;
#pragma unroll
                for (int j = 0; j < 8; ++j) { bf[j] = q[j]; bf[8 + j] = q[16 + j]; }
                acc2a = __builtin_amdgcn_wmma_f32_16x16x32_f16(false, a2, false, bf,
                                                               (short)0, acc2a, false, false);
            }
            {
                const _Float16* q = w2base + (size_t)((ct0 + 1) * 16 + lmod) * Hh + ka;
                v16h bf;
#pragma unroll
                for (int j = 0; j < 8; ++j) { bf[j] = q[j]; bf[8 + j] = q[16 + j]; }
                acc2b = __builtin_amdgcn_wmma_f32_16x16x32_f16(false, a2, false, bf,
                                                               (short)0, acc2b, false, false);
            }
        }
        {
            float bia0 = b2[et * Hh + ct0 * 16 + lmod];
            float bia1 = b2[et * Hh + (ct0 + 1) * 16 + lmod];
#pragma unroll
            for (int j = 0; j < 8; ++j) {
                int m   = j + 8 * lhalf;
                int rr  = r0 + m;
                int bb  = rr / Ee;
                int ee  = rr - bb * Ee;
                float rv = rel[(((size_t)bb * Tt + t) * Ee + ee) * Kk + et] * (1.0f / 3.0f);
                accAll0[j] += fast_tanh(acc2a[j] + bia0) * rv;
                accAll1[j] += fast_tanh(acc2b[j] + bia1) * rv;
            }
        }
    }
    // D layout: element j of lane -> (row = j + 8*(lane>>4), col = lane&15)
#pragma unroll
    for (int j = 0; j < 8; ++j) {
        int m  = j + 8 * lhalf;
        int rr = r0 + m;
        all_msgs[(size_t)rr * Hh + ct0 * 16 + lmod]       = accAll0[j];
        all_msgs[(size_t)rr * Hh + (ct0 + 1) * 16 + lmod] = accAll1[j];
    }
}

// ---------------------------------------------------------------------------
// Aggregate: agg[b][n][h] = (1/(N-1)) * sum over senders s != n of msgs at
// edge e = s*(N-1) + (n < s ? n : n-1). Pure gather, no atomics.
// ---------------------------------------------------------------------------
__global__ __launch_bounds__(256)
void k_agg(const float* __restrict__ all_msgs, float* __restrict__ agg) {
    int blk = blockIdx.x;
    int b = blk / Nn;
    int n = blk - b * Nn;
    int h = threadIdx.x;
    float s = 0.0f;
    for (int snd = 0; snd < Nn; ++snd) {
        if (snd == n) continue;
        int c = (n < snd) ? n : (n - 1);
        int e = snd * (Nn - 1) + c;
        s += all_msgs[((size_t)b * Ee + e) * Hh + h];
    }
    agg[(size_t)(b * Nn + n) * Hh + h] = s * (1.0f / (float)(Nn - 1));
}

// ---------------------------------------------------------------------------
// GRU gate update + 3-layer output MLP for one (b, n) row per block.
// ---------------------------------------------------------------------------
__global__ __launch_bounds__(256)
void k_gru(const float* __restrict__ agg,
           float* __restrict__ insbuf,
           const float* __restrict__ w_hr, const float* __restrict__ w_hi,
           const float* __restrict__ w_hh,
           const float* __restrict__ w_ir, const float* __restrict__ b_ir,
           const float* __restrict__ w_ii, const float* __restrict__ b_ii,
           const float* __restrict__ w_in, const float* __restrict__ b_in,
           const float* __restrict__ w_o1, const float* __restrict__ b_o1,
           const float* __restrict__ w_o2, const float* __restrict__ b_o2,
           const float* __restrict__ w_o3, const float* __restrict__ b_o3,
           float* __restrict__ hidden, _Float16* __restrict__ h16,
           float* __restrict__ out, int t) {
    __shared__ float sAgg[Hh];
    __shared__ float sA[Hh];
    __shared__ float sB[Hh];
    __shared__ float sIns[INN];

    int blk = blockIdx.x;
    int b = blk / Nn;
    int n = blk - b * Nn;
    int h = threadIdx.x;
    size_t rowH = (size_t)(b * Nn + n) * Hh;

    sAgg[h] = agg[rowH + h];
    if (h < INN) sIns[h] = insbuf[(b * Nn + n) * INN + h];
    __syncthreads();

    float ar = 0.0f, ai = 0.0f, an = 0.0f;
    const float* whr = w_hr + (size_t)h * Hh;
    const float* whi = w_hi + (size_t)h * Hh;
    const float* whh = w_hh + (size_t)h * Hh;
    for (int j = 0; j < Hh; ++j) {
        float av = sAgg[j];
        ar += av * whr[j];
        ai += av * whi[j];
        an += av * whh[j];
    }
    float xr = 0.0f, xi = 0.0f, xn = 0.0f;
    for (int i = 0; i < INN; ++i) {
        float v = sIns[i];
        xr += v * w_ir[h * INN + i];
        xi += v * w_ii[h * INN + i];
        xn += v * w_in[h * INN + i];
    }
    float rg = fast_sigmoid(xr + b_ir[h] + ar);
    float ig = fast_sigmoid(xi + b_ii[h] + ai);
    float ng = fast_tanh(xn + b_in[h] + rg * an);
    float hold = hidden[rowH + h];
    float hnew = (1.0f - ig) * ng + ig * hold;
    hidden[rowH + h] = hnew;
    h16[rowH + h] = (_Float16)hnew;
    sA[h] = hnew;
    __syncthreads();

    float p1 = b_o1[h];
    const float* wo1 = w_o1 + (size_t)h * Hh;
    for (int j = 0; j < Hh; ++j) p1 += sA[j] * wo1[j];
    p1 = fmaxf(p1, 0.0f);
    sB[h] = p1;
    __syncthreads();

    float p2 = b_o2[h];
    const float* wo2 = w_o2 + (size_t)h * Hh;
    for (int j = 0; j < Hh; ++j) p2 += sB[j] * wo2[j];
    p2 = fmaxf(p2, 0.0f);
    __syncthreads();       // protect sA before overwrite
    sA[h] = p2;
    __syncthreads();

    if (h < INN) {
        float o = b_o3[h];
        const float* wo3 = w_o3 + (size_t)h * Hh;
        for (int j = 0; j < Hh; ++j) o += sA[j] * wo3[j];
        float pred = sIns[h] + o;
        out[(((size_t)b * Tt + t) * Nn + n) * INN + h] = pred;
        insbuf[(b * Nn + n) * INN + h] = pred;   // next step's ins
    }
}

// ---------------------------------------------------------------------------
extern "C" void kernel_launch(void* const* d_in, const int* in_sizes, int n_in,
                              void* d_out, int out_size, void* d_ws, size_t ws_size,
                              hipStream_t stream) {
    const float* inputs  = (const float*)d_in[0];
    const float* edges   = (const float*)d_in[1];
    const float* msg_w1  = (const float*)d_in[2];
    const float* msg_b1  = (const float*)d_in[3];
    const float* msg_w2  = (const float*)d_in[4];
    const float* msg_b2  = (const float*)d_in[5];
    const float* w_hr    = (const float*)d_in[6];
    const float* w_hi    = (const float*)d_in[7];
    const float* w_hh    = (const float*)d_in[8];
    const float* w_ir    = (const float*)d_in[9];
    const float* b_ir    = (const float*)d_in[10];
    const float* w_ii    = (const float*)d_in[11];
    const float* b_ii    = (const float*)d_in[12];
    const float* w_in    = (const float*)d_in[13];
    const float* b_in    = (const float*)d_in[14];
    const float* w_o1    = (const float*)d_in[15];
    const float* b_o1    = (const float*)d_in[16];
    const float* w_o2    = (const float*)d_in[17];
    const float* b_o2    = (const float*)d_in[18];
    const float* w_o3    = (const float*)d_in[19];
    const float* b_o3    = (const float*)d_in[20];
    float* out = (float*)d_out;

    // Workspace carve (aligned to 256B)
    char* p = (char*)d_ws;
    auto carve = [&](size_t bytes) -> void* {
        void* r = (void*)p;
        p += (bytes + 255) & ~(size_t)255;
        return r;
    };
    float*    all_msgs = (float*)   carve((size_t)BE * Hh * sizeof(float));
    float*    hidden   = (float*)   carve((size_t)Bb * Nn * Hh * sizeof(float));
    _Float16* h16      = (_Float16*)carve((size_t)Bb * Nn * Hh * sizeof(_Float16));
    float*    agg      = (float*)   carve((size_t)Bb * Nn * Hh * sizeof(float));
    float*    insbuf   = (float*)   carve((size_t)Bb * Nn * INN * sizeof(float));
    _Float16* w1h      = (_Float16*)carve((size_t)Kk * Hh * 2 * Hh * sizeof(_Float16));
    _Float16* w2h      = (_Float16*)carve((size_t)Kk * Hh * Hh * sizeof(_Float16));

    {
        int nmax = Bb * Nn * Hh;
        k_init<<<(nmax + 255) / 256, 256, 0, stream>>>(inputs, hidden, h16, insbuf);
    }
    {
        int nmax = Kk * Hh * 2 * Hh;
        k_cvtw<<<(nmax + 255) / 256, 256, 0, stream>>>(msg_w1, msg_w2, w1h, w2h);
    }

    for (int t = 0; t < Tt; ++t) {
        k_msg<<<BE / 16, 256, 0, stream>>>(h16, w1h, msg_b1, w2h, msg_b2,
                                           edges, t, all_msgs);
        k_agg<<<Bb * Nn, 256, 0, stream>>>(all_msgs, agg);
        k_gru<<<Bb * Nn, 256, 0, stream>>>(agg, insbuf,
                                           w_hr, w_hi, w_hh,
                                           w_ir, b_ir, w_ii, b_ii, w_in, b_in,
                                           w_o1, b_o1, w_o2, b_o2, w_o3, b_o3,
                                           hidden, h16, out, t);
    }
}
